// FFTMultiHeadSelfAttention_83279415869578
// MI455X (gfx1250) — compile-verified, matrix operand order checked by reading, not run on hardware
//
#include <hip/hip_runtime.h>
#include <hip/hip_bf16.h>
#include <stdint.h>

// Problem constants (from reference): B=2, S=2048, D=1024, H=16, HD=64
#define BDIM   2
#define SEQ    2048
#define DMODEL 1024
#define NHEAD  16
#define HDIM   64
#define MROWS  (BDIM * SEQ)      // 4096 token rows
#define LN_EPS 1e-5f

typedef __attribute__((ext_vector_type(16))) __bf16       v16bf;
typedef __attribute__((ext_vector_type(8)))  float        v8f;
typedef __attribute__((ext_vector_type(4)))  unsigned int u32x4;
typedef __attribute__((ext_vector_type(4)))  int          v4i;

union FragAB { v16bf v; u32x4 q[2]; };

// gfx1250 async global->LDS copy (ASYNCcnt-tracked), if this toolchain exposes it
#if __has_builtin(__builtin_amdgcn_global_load_async_to_lds_b128) && \
    __has_builtin(__builtin_amdgcn_s_wait_asynccnt)
#define HAVE_ASYNC_LDS 1
#else
#define HAVE_ASYNC_LDS 0
#endif

__device__ __forceinline__ void stage16(const unsigned short* g, unsigned short* l) {
#if HAVE_ASYNC_LDS
    // signature (from hipcc diagnostic): (v4i addrspace(1)*, v4i addrspace(3)*, Ii, Ii)
    __builtin_amdgcn_global_load_async_to_lds_b128(
        (__attribute__((address_space(1))) v4i*)g,
        (__attribute__((address_space(3))) v4i*)l, 0, 0);
#else
    *reinterpret_cast<u32x4*>(l) = *reinterpret_cast<const u32x4*>(g);
#endif
}

// Wait until at most N async copies remain in flight (0 = all landed).
template <int N>
__device__ __forceinline__ void stage_wait() {
#if HAVE_ASYNC_LDS
    __builtin_amdgcn_s_wait_asynccnt(N);
#endif
}

__device__ __forceinline__ unsigned short f2bf(float f) {
    unsigned int u = __float_as_uint(f);
    unsigned int r = (u + 0x7FFFu + ((u >> 16) & 1u)) >> 16;  // RNE
    return (unsigned short)r;
}

// ---------------------------------------------------------------- fp32->bf16
__global__ void cvt_f32_bf16_kernel(const float* __restrict__ in,
                                    unsigned short* __restrict__ out, int n) {
    int i = blockIdx.x * blockDim.x + threadIdx.x;
    if (i < n) out[i] = f2bf(in[i]);
}

// ---------------------------------------------------------- WMMA GEMM (bf16)
// out[M,N] = A[M,K](bf16) @ W[N,K]^T(bf16) + bias[N]   (torch Linear layout)
// 256 threads = 8 waves. Block tile 128x128; A/W 32-wide k-slabs double-buffered
// in LDS via async global->LDS copies; each wave computes a 16x128 strip
// (8 WMMAs / k-step) while the next slab streams in.
template <bool STORE_BF16, bool STORE_F32>
__global__ __launch_bounds__(256) void gemm_bf16_kernel(
    const unsigned short* __restrict__ A,
    const unsigned short* __restrict__ W,
    const float* __restrict__ bias,
    unsigned short* __restrict__ outB,
    float* __restrict__ outF,
    int M, int N, int K)
{
    __shared__ alignas(16) unsigned short lA[2][128 * 32];   // 2 x 8 KiB
    __shared__ alignas(16) unsigned short lW[2][128 * 32];   // 2 x 8 KiB

    const int t    = threadIdx.x;
    const int lane = t & 31;
    const int wave = t >> 5;
    const int half = lane >> 4;
    const int l15  = lane & 15;

    const int rowBase0 = blockIdx.y * 128;
    const int colBase0 = blockIdx.x * 128;
    const int rowBase  = rowBase0 + wave * 16;

    // staging map: thread t -> row t>>1 (0..127), 16-element column group (t&1)*16
    const int sRow = t >> 1;
    const int sC0  = (t & 1) * 16;
    const int sOff = sRow * 32 + sC0;

    v8f zero = {};
    v8f acc[8];
#pragma unroll
    for (int i = 0; i < 8; ++i) acc[i] = zero;

    const size_t gAStage = (size_t)(rowBase0 + sRow) * K + (size_t)sC0;
    const size_t gWStage = (size_t)(colBase0 + sRow) * K + (size_t)sC0;

    // prologue: stage k-slab 0 into buffer 0 (4 async b128 per thread)
    {
        const unsigned short* gA = A + gAStage;
        const unsigned short* gW = W + gWStage;
        stage16(gA,     &lA[0][sOff]);
        stage16(gA + 8, &lA[0][sOff + 8]);
        stage16(gW,     &lW[0][sOff]);
        stage16(gW + 8, &lW[0][sOff + 8]);
    }

    for (int k0 = 0; k0 < K; k0 += 32) {
        const int cur = (k0 >> 5) & 1;
        if (k0 + 32 < K) {
            // stage next slab into the other buffer, then wait only for the
            // current slab (<=4 newest copies may still be in flight)
            const unsigned short* gA = A + gAStage + (k0 + 32);
            const unsigned short* gW = W + gWStage + (k0 + 32);
            stage16(gA,     &lA[cur ^ 1][sOff]);
            stage16(gA + 8, &lA[cur ^ 1][sOff + 8]);
            stage16(gW,     &lW[cur ^ 1][sOff]);
            stage16(gW + 8, &lW[cur ^ 1][sOff + 8]);
            stage_wait<4>();
        } else {
            stage_wait<0>();
        }
        __syncthreads();

        // ---- A fragment for this wave's 16 rows
        FragAB af;
        const unsigned short* ap = &lA[cur][(wave * 16 + l15) * 32 + half * 8];
        af.q[0] = *reinterpret_cast<const u32x4*>(ap);
        af.q[1] = *reinterpret_cast<const u32x4*>(ap + 16);

#pragma unroll
        for (int nt = 0; nt < 8; ++nt) {
            const unsigned short* bp = &lW[cur][(nt * 16 + l15) * 32 + half * 8];
            FragAB bf_;
            bf_.q[0] = *reinterpret_cast<const u32x4*>(bp);
            bf_.q[1] = *reinterpret_cast<const u32x4*>(bp + 16);
            acc[nt] = __builtin_amdgcn_wmma_f32_16x16x32_bf16(
                false, af.v, false, bf_.v, (short)0, acc[nt], false, false);
        }
        __syncthreads();   // all waves done with buf[cur] before it's restaged
    }

    // C layout: VGPR r, lanes 0-15 -> row r, lanes 16-31 -> row r+8; col = l&15
#pragma unroll
    for (int nt = 0; nt < 8; ++nt) {
        const int col = colBase0 + nt * 16 + l15;
        const float bv = bias[col];
#pragma unroll
        for (int r = 0; r < 8; ++r) {
            const int row = rowBase + r + half * 8;
            const float v = acc[nt][r] + bv;
            if constexpr (STORE_BF16) outB[(size_t)row * N + col] = f2bf(v);
            if constexpr (STORE_F32)  outF[(size_t)row * N + col] = v;
        }
    }
}

// --------------------------------------------------- V -> V^T per (b,h) head
// vt[b][h][hd][s] = v[b*S+s][h*HD+hd]; makes the PV contraction contiguous.
__global__ void transpose_v_kernel(const unsigned short* __restrict__ v,
                                   unsigned short* __restrict__ vt) {
    int idx = blockIdx.x * blockDim.x + threadIdx.x;   // 2^22 elements
    int s  = idx & (SEQ - 1);
    int hd = (idx >> 11) & (HDIM - 1);
    int h  = (idx >> 17) & (NHEAD - 1);
    int b  = idx >> 21;
    vt[idx] = v[(size_t)(b * SEQ + s) * DMODEL + h * HDIM + hd];
}

// --------------------------------------------------------- flash attention
// Block = (b, h, 128-query slab): 8 waves x 16 query rows. K (32x64) and V^T
// (64x32) tiles double-buffered in LDS via async copies, shared by all waves.
__global__ __launch_bounds__(256) void attn_kernel(
    const unsigned short* __restrict__ Q,
    const unsigned short* __restrict__ Km,
    const unsigned short* __restrict__ Vt,
    unsigned short* __restrict__ Ctx)
{
    __shared__ alignas(16) unsigned short lK[2][32 * 64];    // 2 x 4 KiB [key][hd]
    __shared__ alignas(16) unsigned short lV[2][64 * 32];    // 2 x 4 KiB [hd][key]
    __shared__ alignas(16) unsigned short pbuf[8][16 * 32];  // 8 KiB per-wave P

    const int t    = threadIdx.x;
    const int lane = t & 31;
    const int wave = t >> 5;
    const int half = lane >> 4;
    const int l15  = lane & 15;

    const int qblk = blockIdx.x & 15;            // 16 slabs of 128 queries
    const int h    = (blockIdx.x >> 4) & (NHEAD - 1);
    const int b    = blockIdx.x >> 8;

    const int qRow0 = b * SEQ + qblk * 128 + wave * 16;
    const int hcol  = h * HDIM;

    // staging maps
    const int kRow = t >> 3, kC = (t & 7) * 8;   // K tile: 32 rows x 64 cols
    const int vRow = t >> 2, vC = (t & 3) * 8;   // V tile: 64 rows x 32 cols
    const int kOff = kRow * 64 + kC;
    const int vOff = vRow * 32 + vC;
    const size_t gK0 = (size_t)(b * SEQ + kRow) * DMODEL + (size_t)(hcol + kC);
    const size_t gV0 = (size_t)((b * NHEAD + h) * HDIM + vRow) * SEQ + (size_t)vC;

    // Q fragments (HD=64 -> two K=32 fragments), loaded once, reused all blocks
    FragAB qf[2];
    {
        const unsigned short* qp =
            Q + (size_t)(qRow0 + l15) * DMODEL + (size_t)(hcol + half * 8);
        qf[0].q[0] = *reinterpret_cast<const u32x4*>(qp);
        qf[0].q[1] = *reinterpret_cast<const u32x4*>(qp + 16);
        qf[1].q[0] = *reinterpret_cast<const u32x4*>(qp + 32);
        qf[1].q[1] = *reinterpret_cast<const u32x4*>(qp + 48);
    }

    v8f zero = {};
    v8f ctx[4];
#pragma unroll
    for (int i = 0; i < 4; ++i) ctx[i] = zero;

    float mrun[8], lrun[8];
#pragma unroll
    for (int r = 0; r < 8; ++r) { mrun[r] = -1e30f; lrun[r] = 0.0f; }

    const float sc = 0.03125f;   // 1/sqrt(D) = 1/32

    // prologue: stage key-block 0 into buffer 0 (2 async b128 per thread)
    stage16(Km + gK0, &lK[0][kOff]);
    stage16(Vt + gV0, &lV[0][vOff]);

    for (int kb = 0; kb < SEQ; kb += 32) {
        const int cur = (kb >> 5) & 1;
        if (kb + 32 < SEQ) {
            stage16(Km + gK0 + (size_t)(kb + 32) * DMODEL, &lK[cur ^ 1][kOff]);
            stage16(Vt + gV0 + (size_t)(kb + 32),          &lV[cur ^ 1][vOff]);
            stage_wait<2>();   // current tile landed; next may still fly
        } else {
            stage_wait<0>();
        }
        __syncthreads();

        // pull the tile after next toward L2 while we compute this one
        if (kb + 64 < SEQ) {
            __builtin_prefetch(Km + gK0 + (size_t)(kb + 64) * DMODEL, 0, 1);
            __builtin_prefetch(Vt + gV0 + (size_t)(kb + 64), 0, 1);
        }

        // ---- scores: two 16x16 tiles (32 keys), contraction HD = 64
        v8f s[2];
        s[0] = zero; s[1] = zero;
#pragma unroll
        for (int jt = 0; jt < 2; ++jt) {
            const unsigned short* kp = &lK[cur][(jt * 16 + l15) * 64 + half * 8];
            FragAB kf;
            kf.q[0] = *reinterpret_cast<const u32x4*>(kp);
            kf.q[1] = *reinterpret_cast<const u32x4*>(kp + 16);
            s[jt] = __builtin_amdgcn_wmma_f32_16x16x32_bf16(
                false, qf[0].v, false, kf.v, (short)0, s[jt], false, false);
            kf.q[0] = *reinterpret_cast<const u32x4*>(kp + 32);
            kf.q[1] = *reinterpret_cast<const u32x4*>(kp + 48);
            s[jt] = __builtin_amdgcn_wmma_f32_16x16x32_bf16(
                false, qf[1].v, false, kf.v, (short)0, s[jt], false, false);
        }

        // ---- online softmax; row r lives in lane-half (rows r / r+8)
#pragma unroll
        for (int r = 0; r < 8; ++r) {
            float a = s[0][r] * sc;
            float c = s[1][r] * sc;
            float mx = fmaxf(a, c);
#pragma unroll
            for (int o = 1; o < 16; o <<= 1) mx = fmaxf(mx, __shfl_xor(mx, o));
            float mnew  = fmaxf(mrun[r], mx);
            float scale = __expf(mrun[r] - mnew);
            float p0 = __expf(a - mnew);
            float p1 = __expf(c - mnew);
            float rs = p0 + p1;
#pragma unroll
            for (int o = 1; o < 16; o <<= 1) rs += __shfl_xor(rs, o);
            lrun[r] = lrun[r] * scale + rs;
            mrun[r] = mnew;
#pragma unroll
            for (int nt = 0; nt < 4; ++nt) ctx[nt][r] = ctx[nt][r] * scale;

            const int row = r + half * 8;
            pbuf[wave][row * 32 + l15]      = f2bf(p0);
            pbuf[wave][row * 32 + 16 + l15] = f2bf(p1);
        }
        // per-wave LDS round-trip: DS ops from one wave complete in order, so
        // a scheduling-only barrier is enough (no workgroup barrier needed).
        __builtin_amdgcn_wave_barrier();

        // ---- P (16x32) as A fragment from LDS
        FragAB pf;
        const unsigned short* pp = &pbuf[wave][l15 * 32 + half * 8];
        pf.q[0] = *reinterpret_cast<const u32x4*>(pp);
        pf.q[1] = *reinterpret_cast<const u32x4*>(pp + 16);

        // ---- ctx += P @ V_block  (N = HD = 64 -> 4 tiles), V^T k-contiguous
#pragma unroll
        for (int nt = 0; nt < 4; ++nt) {
            const unsigned short* vp = &lV[cur][(nt * 16 + l15) * 32 + half * 8];
            FragAB vf;
            vf.q[0] = *reinterpret_cast<const u32x4*>(vp);
            vf.q[1] = *reinterpret_cast<const u32x4*>(vp + 16);
            ctx[nt] = __builtin_amdgcn_wmma_f32_16x16x32_bf16(
                false, pf.v, false, vf.v, (short)0, ctx[nt], false, false);
        }
        __syncthreads();   // all waves done with buf[cur] before it's restaged
    }

    // ---- finalize: ctx / l, store bf16 into [4096, 1024] ctx buffer
#pragma unroll
    for (int nt = 0; nt < 4; ++nt) {
#pragma unroll
        for (int r = 0; r < 8; ++r) {
            const int row = qRow0 + r + half * 8;
            const int col = hcol + nt * 16 + l15;
            Ctx[(size_t)row * DMODEL + col] = f2bf(ctx[nt][r] / lrun[r]);
        }
    }
}

// ------------------------------------------------ residual add + LayerNorm
__global__ __launch_bounds__(256) void add_ln_kernel(
    const float* __restrict__ x, const float* __restrict__ resid,
    const float* __restrict__ gamma, const float* __restrict__ beta,
    float* __restrict__ out)
{
    const int row  = blockIdx.x;
    const int t    = threadIdx.x;
    const int lane = t & 31;
    const int wave = t >> 5;
    __shared__ float s1a[8], s2a[8];

    float vals[4];
    float s1 = 0.f, s2 = 0.f;
#pragma unroll
    for (int i = 0; i < 4; ++i) {
        const int c = t + i * 256;
        const float v = x[(size_t)row * DMODEL + c] + resid[(size_t)row * DMODEL + c];
        vals[i] = v; s1 += v; s2 += v * v;
    }
#pragma unroll
    for (int o = 1; o < 32; o <<= 1) { s1 += __shfl_xor(s1, o); s2 += __shfl_xor(s2, o); }
    if (lane == 0) { s1a[wave] = s1; s2a[wave] = s2; }
    __syncthreads();
    s1 = 0.f; s2 = 0.f;
#pragma unroll
    for (int w = 0; w < 8; ++w) { s1 += s1a[w]; s2 += s2a[w]; }
    const float mu   = s1 * (1.0f / DMODEL);
    const float var  = s2 * (1.0f / DMODEL) - mu * mu;
    const float rstd = rsqrtf(var + LN_EPS);
#pragma unroll
    for (int i = 0; i < 4; ++i) {
        const int c = t + i * 256;
        out[(size_t)row * DMODEL + c] = (vals[i] - mu) * rstd * gamma[c] + beta[c];
    }
}

// ---------------------------------------------------------------- launcher
extern "C" void kernel_launch(void* const* d_in, const int* in_sizes, int n_in,
                              void* d_out, int out_size, void* d_ws, size_t ws_size,
                              hipStream_t stream) {
    (void)in_sizes; (void)n_in; (void)out_size; (void)ws_size;

    const float* x     = (const float*)d_in[0];
    const float* wq    = (const float*)d_in[1];
    const float* bq    = (const float*)d_in[2];
    const float* wk    = (const float*)d_in[3];
    const float* bk    = (const float*)d_in[4];
    const float* wv    = (const float*)d_in[5];
    const float* bv    = (const float*)d_in[6];
    const float* wo    = (const float*)d_in[7];
    const float* bo    = (const float*)d_in[8];
    const float* gamma = (const float*)d_in[9];
    const float* beta  = (const float*)d_in[10];
    float* out = (float*)d_out;

    char* ws = (char*)d_ws;
    unsigned short* xb   = (unsigned short*)(ws);                       //  8 MiB
    unsigned short* wqb  = (unsigned short*)(ws + (size_t)( 8u << 20)); //  2 MiB
    unsigned short* wkb  = (unsigned short*)(ws + (size_t)(10u << 20));
    unsigned short* wvb  = (unsigned short*)(ws + (size_t)(12u << 20));
    unsigned short* wob  = (unsigned short*)(ws + (size_t)(14u << 20));
    unsigned short* qb   = (unsigned short*)(ws + (size_t)(16u << 20)); //  8 MiB
    unsigned short* kb   = (unsigned short*)(ws + (size_t)(24u << 20));
    unsigned short* vb   = (unsigned short*)(ws + (size_t)(32u << 20));
    unsigned short* vtb  = (unsigned short*)(ws + (size_t)(40u << 20));
    unsigned short* ctxb = (unsigned short*)(ws + (size_t)(48u << 20));
    float*          resid = (float*)(ws + (size_t)(56u << 20));          // 16 MiB

    const int NX = MROWS * DMODEL;    // 4194304
    const int NW = DMODEL * DMODEL;   // 1048576

    cvt_f32_bf16_kernel<<<NX / 256, 256, 0, stream>>>(x,  xb,  NX);
    cvt_f32_bf16_kernel<<<NW / 256, 256, 0, stream>>>(wq, wqb, NW);
    cvt_f32_bf16_kernel<<<NW / 256, 256, 0, stream>>>(wk, wkb, NW);
    cvt_f32_bf16_kernel<<<NW / 256, 256, 0, stream>>>(wv, wvb, NW);
    cvt_f32_bf16_kernel<<<NW / 256, 256, 0, stream>>>(wo, wob, NW);

    dim3 gGemm(DMODEL / 128, MROWS / 128);
    gemm_bf16_kernel<true, false><<<gGemm, 256, 0, stream>>>(
        xb, wqb, bq, qb, nullptr, MROWS, DMODEL, DMODEL);
    gemm_bf16_kernel<true, false><<<gGemm, 256, 0, stream>>>(
        xb, wkb, bk, kb, nullptr, MROWS, DMODEL, DMODEL);
    gemm_bf16_kernel<true, false><<<gGemm, 256, 0, stream>>>(
        xb, wvb, bv, vb, nullptr, MROWS, DMODEL, DMODEL);

    transpose_v_kernel<<<NX / 256, 256, 0, stream>>>(vb, vtb);

    attn_kernel<<<512, 256, 0, stream>>>(qb, kb, vtb, ctxb);

    gemm_bf16_kernel<false, true><<<gGemm, 256, 0, stream>>>(
        ctxb, wob, bo, nullptr, resid, MROWS, DMODEL, DMODEL);

    add_ln_kernel<<<MROWS, 256, 0, stream>>>(x, resid, gamma, beta, out);
}
